// FlaxDebertaV2Attention_6468220748569
// MI455X (gfx1250) — matrix operand order repaired and checked by reading, not compiled, not run on hardware
//
#include <hip/hip_runtime.h>
#include <hip/hip_bf16.h>
#include <math.h>

typedef __attribute__((ext_vector_type(16))) _Float16 v16h;
typedef __attribute__((ext_vector_type(8)))  float    v8f;

#define NB   4
#define SEQ  1024
#define HID  1024
#define NH   16
#define DH   64
#define SPAN 256          // ATT_SPAN
#define P2   (2 * SPAN)   // 512
#define TBL  2047

// padded LDS strides (all 16B-aligned, not 0 mod 64 banks)
#define SCST  1032   // score strip stride (f32)
#define CPST  520    // c2p strip stride (f32)
#define P16ST 1040   // prob strip stride (f16)
#define QKST  72     // q/k/v tile stride (f16)
#define KSZ   (64 * QKST)   // halves per K/V tile buffer
#define GST   40     // gemm tile stride (f16)

// ---------------------------------------------------------------------------
// CDNA5 async global->LDS copy (ASYNCcnt-tracked), 16 bytes per lane.
// LDS byte address = low 32 bits of the generic pointer (aperture rule).
// ---------------------------------------------------------------------------
__device__ __forceinline__ void async_copy_b128(void* lds_dst, const void* gsrc) {
  unsigned lds_addr = (unsigned)(uintptr_t)lds_dst;
  asm volatile("global_load_async_to_lds_b128 %0, %1, off"
               :: "v"(lds_addr), "v"(gsrc) : "memory");
}
__device__ __forceinline__ void async_wait0() {
  asm volatile("s_wait_asynccnt 0" ::: "memory");
}
__device__ __forceinline__ void async_wait4() {  // ops retire in order
  asm volatile("s_wait_asynccnt 4" ::: "memory");
}

// ---------------------------------------------------------------------------
// WMMA helpers (gfx1250, wave32)
// ---------------------------------------------------------------------------

// A fragment, f16 16x32 (MxK), row-major source: caller passes pointer at
// (row = lane%16, K=0). lanes 0-15 hold K {0..7,16..23}, lanes 16-31 hold
// K {8..15,24..31} (ISA 7.12.2).
__device__ __forceinline__ v16h load_frag_a(const _Float16* p, int lane) {
  const int hi = (lane >> 4) * 8;
  v16h f;
#pragma unroll
  for (int t = 0; t < 4; ++t) {
    f[2 * t]     = p[hi + 2 * t];
    f[2 * t + 1] = p[hi + 2 * t + 1];
    f[8 + 2 * t]     = p[16 + hi + 2 * t];
    f[8 + 2 * t + 1] = p[16 + hi + 2 * t + 1];
  }
  return f;
}

// B fragment, f16 32x16 (KxN): lane holds column N = lane%16; lanes 0-15 hold
// K {0..7,16..23}, lanes 16-31 hold K {8..15,24..31}. K contiguous at *p.
__device__ __forceinline__ v16h load_frag_b(const _Float16* p, int lane) {
  p += (lane >> 4) * 8;
  v16h f;
#pragma unroll
  for (int t = 0; t < 8; ++t) { f[t] = p[t]; f[8 + t] = p[16 + t]; }
  return f;
}

__device__ __forceinline__ v8f wmma16(v16h a, v16h b, v8f c) {
  return __builtin_amdgcn_wmma_f32_16x16x32_f16(false, a, false, b, (short)0, c,
                                                false, false);
}

// ---------------------------------------------------------------------------
// Log-bucket relative-position tables
// ---------------------------------------------------------------------------
__global__ void build_tables_kernel(int* __restrict__ tbl_c2p,
                                    int* __restrict__ tbl_p2c) {
  int idx = blockIdx.x * blockDim.x + threadIdx.x;
  if (idx >= TBL) return;
  int rel = idx - 1023;
  int sgn = (rel > 0) - (rel < 0);
  int abs_pos = (rel < 128 && rel > -128) ? 127 : (rel < 0 ? -rel : rel);
  int bucket;
  if (abs_pos <= 128) {
    bucket = abs_pos;
  } else {
    float lp = ceilf(logf((float)abs_pos * (1.0f / 128.0f)) /
                     logf(511.0f / 128.0f) * 127.0f) + 128.0f;
    bucket = (int)lp * sgn;
  }
  int c = bucket + 256;  c = c < 0 ? 0 : (c > 511 ? 511 : c);
  int p = -bucket + 256; p = p < 0 ? 0 : (p > 511 ? 511 : p);
  tbl_c2p[idx] = c;
  tbl_p2c[idx] = p;
}

__global__ void cvt_f32_f16_kernel(const float* __restrict__ in,
                                   _Float16* __restrict__ out, int n) {
  int i = blockIdx.x * blockDim.x + threadIdx.x;
  int stride = gridDim.x * blockDim.x;
  for (; i < n; i += stride) out[i] = (_Float16)in[i];
}

// ---------------------------------------------------------------------------
// Generic WMMA GEMM, double-buffered:
//   out[f16 MxN] = A[f16 MxK] @ B[f16 KxN] + bias[f32 N]
// 64x64 tile / workgroup, 128 threads (4 waves). A tile via async DMA into
// ping-pong buffers; B tile via global->register loads (issued before
// compute) + transpose ds-stores (after compute).
// ---------------------------------------------------------------------------
__global__ void gemm_bias_f16_kernel(const _Float16* __restrict__ A,
                                     const _Float16* __restrict__ Bm,
                                     const float* __restrict__ bias,
                                     _Float16* __restrict__ out,
                                     int M, int N, int K) {
  __shared__ _Float16 a_s[2][64 * GST];
  __shared__ _Float16 bt_s[2][64 * GST];
  const int tid = threadIdx.x, lane = tid & 31, wid = tid >> 5;
  const int m0 = blockIdx.x * 64, n0 = blockIdx.y * 64;
  const int arow = tid >> 2, acol = tid & 3;   // A chunks: rows arow, arow+32
  const int bkk = tid >> 3, bcol = tid & 7;    // B chunks: kk bkk, bkk+16
  v8f zero = {};
  v8f acc[4];
  acc[0] = zero; acc[1] = zero; acc[2] = zero; acc[3] = zero;

  auto stageA = [&](int k0, int buf) {
    async_copy_b128(&a_s[buf][arow * GST + acol * 8],
                    A + (size_t)(m0 + arow) * K + k0 + acol * 8);
    async_copy_b128(&a_s[buf][(arow + 32) * GST + acol * 8],
                    A + (size_t)(m0 + arow + 32) * K + k0 + acol * 8);
  };
  auto loadB = [&](int k0, uint4& r0, uint4& r1) {
    r0 = *(const uint4*)(Bm + (size_t)(k0 + bkk) * N + n0 + bcol * 8);
    r1 = *(const uint4*)(Bm + (size_t)(k0 + bkk + 16) * N + n0 + bcol * 8);
  };
  auto storeB = [&](int buf, uint4 r0, uint4 r1) {
    alignas(16) _Float16 t0[8], t1[8];
    *(uint4*)t0 = r0;
    *(uint4*)t1 = r1;
#pragma unroll
    for (int j = 0; j < 8; ++j) {
      bt_s[buf][(bcol * 8 + j) * GST + bkk]      = t0[j];
      bt_s[buf][(bcol * 8 + j) * GST + bkk + 16] = t1[j];
    }
  };

  // prologue: tile 0
  uint4 br0, br1;
  stageA(0, 0);
  loadB(0, br0, br1);
  storeB(0, br0, br1);
  async_wait0();
  __syncthreads();

  const int nsteps = K / 32;
  for (int t = 0; t < nsteps; ++t) {
    const int cur = t & 1, nxt = cur ^ 1;
    const bool more = (t + 1 < nsteps);
    if (more) {
      stageA((t + 1) * 32, nxt);   // DMA overlaps compute below
      loadB((t + 1) * 32, br0, br1);
    }
    v16h af = load_frag_a(&a_s[cur][(wid * 16 + (lane & 15)) * GST], lane);
#pragma unroll
    for (int n = 0; n < 4; ++n) {
      v16h bf = load_frag_b(&bt_s[cur][(n * 16 + (lane & 15)) * GST], lane);
      acc[n] = wmma16(af, bf, acc[n]);
    }
    if (more) storeB(nxt, br0, br1);
    async_wait0();
    __syncthreads();
  }

  const int cbase = n0 + (lane & 15);
  const int rbase = m0 + wid * 16 + ((lane >> 4) * 8);
#pragma unroll
  for (int n = 0; n < 4; ++n) {
    int cg = cbase + n * 16;
    float bv = bias[cg];
#pragma unroll
    for (int i = 0; i < 8; ++i)
      out[(size_t)(rbase + i) * N + cg] = (_Float16)(acc[n][i] + bv);
  }
}

// Output projection: preln[f32] = ctx @ Wo + bo + hidden (residual)
__global__ void gemm_out_res_kernel(const _Float16* __restrict__ A,
                                    const _Float16* __restrict__ Bm,
                                    const float* __restrict__ bias,
                                    const float* __restrict__ hidden,
                                    float* __restrict__ preln,
                                    int M, int N, int K) {
  __shared__ _Float16 a_s[2][64 * GST];
  __shared__ _Float16 bt_s[2][64 * GST];
  const int tid = threadIdx.x, lane = tid & 31, wid = tid >> 5;
  const int m0 = blockIdx.x * 64, n0 = blockIdx.y * 64;
  const int arow = tid >> 2, acol = tid & 3;
  const int bkk = tid >> 3, bcol = tid & 7;
  v8f zero = {};
  v8f acc[4];
  acc[0] = zero; acc[1] = zero; acc[2] = zero; acc[3] = zero;

  auto stageA = [&](int k0, int buf) {
    async_copy_b128(&a_s[buf][arow * GST + acol * 8],
                    A + (size_t)(m0 + arow) * K + k0 + acol * 8);
    async_copy_b128(&a_s[buf][(arow + 32) * GST + acol * 8],
                    A + (size_t)(m0 + arow + 32) * K + k0 + acol * 8);
  };
  auto loadB = [&](int k0, uint4& r0, uint4& r1) {
    r0 = *(const uint4*)(Bm + (size_t)(k0 + bkk) * N + n0 + bcol * 8);
    r1 = *(const uint4*)(Bm + (size_t)(k0 + bkk + 16) * N + n0 + bcol * 8);
  };
  auto storeB = [&](int buf, uint4 r0, uint4 r1) {
    alignas(16) _Float16 t0[8], t1[8];
    *(uint4*)t0 = r0;
    *(uint4*)t1 = r1;
#pragma unroll
    for (int j = 0; j < 8; ++j) {
      bt_s[buf][(bcol * 8 + j) * GST + bkk]      = t0[j];
      bt_s[buf][(bcol * 8 + j) * GST + bkk + 16] = t1[j];
    }
  };

  uint4 br0, br1;
  stageA(0, 0);
  loadB(0, br0, br1);
  storeB(0, br0, br1);
  async_wait0();
  __syncthreads();

  const int nsteps = K / 32;
  for (int t = 0; t < nsteps; ++t) {
    const int cur = t & 1, nxt = cur ^ 1;
    const bool more = (t + 1 < nsteps);
    if (more) {
      stageA((t + 1) * 32, nxt);
      loadB((t + 1) * 32, br0, br1);
    }
    v16h af = load_frag_a(&a_s[cur][(wid * 16 + (lane & 15)) * GST], lane);
#pragma unroll
    for (int n = 0; n < 4; ++n) {
      v16h bf = load_frag_b(&bt_s[cur][(n * 16 + (lane & 15)) * GST], lane);
      acc[n] = wmma16(af, bf, acc[n]);
    }
    if (more) storeB(nxt, br0, br1);
    async_wait0();
    __syncthreads();
  }

  const int cbase = n0 + (lane & 15);
  const int rbase = m0 + wid * 16 + ((lane >> 4) * 8);
#pragma unroll
  for (int n = 0; n < 4; ++n) {
    int cg = cbase + n * 16;
    float bv = bias[cg];
#pragma unroll
    for (int i = 0; i < 8; ++i) {
      size_t off = (size_t)(rbase + i) * N + cg;
      preln[off] = acc[n][i] + bv + hidden[off];
    }
  }
}

// ---------------------------------------------------------------------------
// p2c_att[b,h,k,p] = sum_d K[b,k,h*64+d] * posk[p,h*64+d]   (fp16 out)
// ---------------------------------------------------------------------------
__global__ void p2c_att_kernel(const _Float16* __restrict__ K16,
                               const _Float16* __restrict__ posk,
                               _Float16* __restrict__ p2c) {
  const int kb = blockIdx.x, pb = blockIdx.y, bh = blockIdx.z;
  const int b = bh >> 4, h = bh & 15;
  const int lane = threadIdx.x & 31, wid = threadIdx.x >> 5;
  const int k0 = kb * 32, p0 = pb * 64;
  v8f zero = {};
#pragma unroll
  for (int f = 0; f < 2; ++f) {
    int idx = wid + f * 4;
    int m = idx >> 2, n = idx & 3;
    v8f acc = zero;
#pragma unroll
    for (int kk = 0; kk < 64; kk += 32) {
      const _Float16* ap =
          K16 + (size_t)(b * SEQ + k0 + m * 16 + (lane & 15)) * HID + h * DH + kk;
      const _Float16* bp =
          posk + (size_t)(p0 + n * 16 + (lane & 15)) * HID + h * DH + kk;
      acc = wmma16(load_frag_a(ap, lane), load_frag_b(bp, lane), acc);
    }
    int kg = k0 + m * 16 + ((lane >> 4) * 8);
    int pg = p0 + n * 16 + (lane & 15);
#pragma unroll
    for (int i = 0; i < 8; ++i)
      p2c[((size_t)bh * SEQ + kg + i) * P2 + pg] = (_Float16)acc[i];
  }
}

// ---------------------------------------------------------------------------
// Fused attention per (b, h, 32-query tile), 128 threads (4 waves),
// ~217 KB dynamic LDS (CDNA5: 320 KB / workgroup). Double-buffered K tiles
// (async DMA overlapping WMMA), register-pipelined V tiles.
// ---------------------------------------------------------------------------
__global__ void attn_kernel(const _Float16* __restrict__ Q16,
                            const _Float16* __restrict__ K16,
                            const _Float16* __restrict__ V16,
                            const _Float16* __restrict__ posk,
                            const _Float16* __restrict__ p2c,
                            const int* __restrict__ tbl_c2p,
                            const int* __restrict__ tbl_p2c,
                            _Float16* __restrict__ ctx) {
  extern __shared__ char smem[];
  float*    sc   = (float*)smem;                 // 32 x SCST f32  (132096 B)
  float*    cp   = (float*)(smem + 132096);      // 32 x CPST f32  (66560 B)
  _Float16* p16  = (_Float16*)(smem + 132096);   // alias: 32 x P16ST f16
  _Float16* qs   = (_Float16*)(smem + 198656);   // 32 x QKST f16  (4608 B)
  _Float16* ks   = (_Float16*)(smem + 203264);   // 2 x 64 x QKST  (18432 B)
  float*    red  = (float*)(smem + 221696);      // 32 x 4
  float*    redm = (float*)(smem + 222208);      // 32

  const int tid = threadIdx.x, lane = tid & 31, wid = tid >> 5;
  const int qb = blockIdx.x, h = blockIdx.y, b = blockIdx.z;
  const int q0 = qb * 32;
  const float inv_scale = 0.07216878364870323f;  // 1/sqrt(64*3)
  const int krow = tid >> 3, kcol = tid & 7;     // 64-row tile chunk coords
  v8f zero = {};

  const _Float16* kbase = K16 + (size_t)(b * SEQ) * HID + h * DH;
  auto stage_k = [&](int kt, int buf) {
    const _Float16* src = kbase + (size_t)(kt * 64) * HID;
#pragma unroll
    for (int i = 0; i < 4; ++i) {
      int r = krow + i * 16;
      async_copy_b128(ks + buf * KSZ + r * QKST + kcol * 8,
                      src + (size_t)r * HID + kcol * 8);
    }
  };

  // ---- Phase A: async-stage Q tile, then prefetch K tile 0 ----
  {
    const _Float16* src = Q16 + (size_t)(b * SEQ + q0) * HID + h * DH;
#pragma unroll
    for (int i = 0; i < 2; ++i) {
      int idx = tid + i * 128;
      int r = idx >> 3, c = idx & 7;
      async_copy_b128(qs + r * QKST + c * 8, src + (size_t)r * HID + c * 8);
    }
  }
  stage_k(0, 0);        // K tile 0 DMA overlaps Phase B compute
  async_wait4();        // in-order retirement: Q's 2 copies are done
  __syncthreads();

  // ---- Phase B: c2p strip [32 x 512] = Q_tile @ posk_h^T ----
  for (int fi = wid; fi < 64; fi += 4) {
    int m = fi & 1, n = fi >> 1;
    v8f acc = zero;
#pragma unroll
    for (int kk = 0; kk < 64; kk += 32) {
      v16h af = load_frag_a(qs + (m * 16 + (lane & 15)) * QKST + kk, lane);
      v16h bf = load_frag_b(
          posk + (size_t)(n * 16 + (lane & 15)) * HID + h * DH + kk, lane);
      acc = wmma16(af, bf, acc);
    }
    int rl = m * 16 + ((lane >> 4) * 8);
    int cl = n * 16 + (lane & 15);
#pragma unroll
    for (int i = 0; i < 8; ++i) cp[(rl + i) * CPST + cl] = acc[i];
  }
  async_wait0();        // K tile 0 landed
  __syncthreads();      // cp + ks[0] visible everywhere

  // ---- Phase C: scores = (QK^T + c2p + p2c) / sqrt(3d), ping-pong K ----
  const size_t p2c_base = (size_t)(b * NH + h) * SEQ;
  for (int kt = 0; kt < 16; ++kt) {
    const int cur = kt & 1;
    if (kt + 1 < 16) stage_k(kt + 1, cur ^ 1);   // DMA overlaps compute
    const int k0 = kt * 64;
    const _Float16* kbuf = ks + cur * KSZ;
#pragma unroll
    for (int m = 0; m < 2; ++m) {
      int n = wid;
      v8f acc = zero;
#pragma unroll
      for (int kk = 0; kk < 64; kk += 32) {
        v16h af = load_frag_a(qs + (m * 16 + (lane & 15)) * QKST + kk, lane);
        v16h bf = load_frag_b(kbuf + (n * 16 + (lane & 15)) * QKST + kk, lane);
        acc = wmma16(af, bf, acc);
      }
      int rl = m * 16 + ((lane >> 4) * 8);
      int kg = k0 + n * 16 + (lane & 15);
#pragma unroll
      for (int i = 0; i < 8; ++i) {
        int qg = q0 + rl + i;
        int dlt = qg - kg + 1023;                       // [0, 2046]
        float c2pv = cp[(rl + i) * CPST + tbl_c2p[dlt]];
        float p2cv = (float)p2c[(p2c_base + kg) * P2 + tbl_p2c[2046 - dlt]];
        sc[(rl + i) * SCST + kg] = (acc[i] + c2pv + p2cv) * inv_scale;
      }
    }
    async_wait0();
    __syncthreads();
  }

  // ---- V tile 0 prefetch into registers (latency hides under softmax) ----
  const _Float16* vbase = V16 + (size_t)(b * SEQ) * HID + h * DH;
  uint4 vreg[4];
#pragma unroll
  for (int i = 0; i < 4; ++i)
    vreg[i] = *(const uint4*)(vbase + (size_t)(krow + i * 16) * HID + kcol * 8);

  // ---- Phase D: exact softmax (4 threads/row, stride-4 cols: bank-clean) ----
  {
    int r = tid >> 2, part = tid & 3;
    float mx = -1e30f;
    for (int j = 0; j < 256; ++j)
      mx = fmaxf(mx, sc[r * SCST + part + 4 * j]);
    red[r * 4 + part] = mx;
    __syncthreads();
    if (tid < 32) {
      float m4 = fmaxf(fmaxf(red[tid * 4], red[tid * 4 + 1]),
                       fmaxf(red[tid * 4 + 2], red[tid * 4 + 3]));
      redm[tid] = m4;
    }
    __syncthreads();
    float rm = redm[r];
    float s = 0.f;
    for (int j = 0; j < 256; ++j) {
      int c = part + 4 * j;
      float e = __expf(sc[r * SCST + c] - rm);
      sc[r * SCST + c] = e;
      s += e;
    }
    red[r * 4 + part] = s;
    __syncthreads();
    if (tid < 32) {
      float s4 = red[tid * 4] + red[tid * 4 + 1] + red[tid * 4 + 2] +
                 red[tid * 4 + 3];
      redm[tid] = 1.0f / s4;
    }
    __syncthreads();
    // normalized probs as packed fp16 pairs into p16 (aliases cp; cp dead)
    for (int e2 = tid; e2 < 32 * 512; e2 += 128) {
      int rr = e2 >> 9;
      int c = (e2 & 511) * 2;
      float inv = redm[rr];
      union { unsigned u; _Float16 hh[2]; } pk;
      pk.hh[0] = (_Float16)(sc[rr * SCST + c] * inv);
      pk.hh[1] = (_Float16)(sc[rr * SCST + c + 1] * inv);
      *(unsigned*)(p16 + rr * P16ST + c) = pk.u;
    }
  }

  // ---- Phase E: ctx = P @ V, register-pipelined V transpose staging ----
  v8f acc0 = zero, acc1 = zero;
  _Float16* vt = ks;  // reuse K buffer 0: V transposed, vt[d][k], stride QKST
  for (int kt = 0; kt < 16; ++kt) {
    const int k0 = kt * 64;
    __syncthreads();  // prior vt reads (and p16 writes on kt==0) complete
#pragma unroll
    for (int i = 0; i < 4; ++i) {
      alignas(16) _Float16 t[8];
      *(uint4*)t = vreg[i];
      int r = krow + i * 16;
#pragma unroll
      for (int j = 0; j < 8; ++j) vt[(kcol * 8 + j) * QKST + r] = t[j];
    }
    if (kt + 1 < 16) {  // next tile global loads overlap compute below
      const _Float16* nb = vbase + (size_t)((kt + 1) * 64) * HID;
#pragma unroll
      for (int i = 0; i < 4; ++i)
        vreg[i] = *(const uint4*)(nb + (size_t)(krow + i * 16) * HID + kcol * 8);
    }
    __syncthreads();
#pragma unroll
    for (int kk = 0; kk < 64; kk += 32) {
      v16h bf = load_frag_b(vt + (wid * 16 + (lane & 15)) * QKST + kk, lane);
      v16h a0 = load_frag_a(p16 + (size_t)(lane & 15) * P16ST + k0 + kk, lane);
      acc0 = wmma16(a0, bf, acc0);
      v16h a1 =
          load_frag_a(p16 + (size_t)(16 + (lane & 15)) * P16ST + k0 + kk, lane);
      acc1 = wmma16(a1, bf, acc1);
    }
  }
  {
    int cl = wid * 16 + (lane & 15);
    int r0 = (lane >> 4) * 8;
    _Float16* dst = ctx + (size_t)(b * SEQ + q0) * HID + h * DH + cl;
#pragma unroll
    for (int i = 0; i < 8; ++i) {
      dst[(size_t)(r0 + i) * HID]      = (_Float16)acc0[i];
      dst[(size_t)(16 + r0 + i) * HID] = (_Float16)acc1[i];
    }
  }
}

// ---------------------------------------------------------------------------
// LayerNorm over last dim (H=1024), one row per workgroup.
// ---------------------------------------------------------------------------
__global__ void ln_kernel(const float* __restrict__ preln,
                          const float* __restrict__ gamma,
                          const float* __restrict__ beta,
                          float* __restrict__ out) {
  __shared__ float rs[8];
  __shared__ float rs2[8];
  const int row = blockIdx.x;
  const float* x = preln + (size_t)row * HID;
  float s = 0.f, s2 = 0.f;
  for (int c = threadIdx.x; c < HID; c += blockDim.x) {
    float v = x[c];
    s += v;
    s2 += v * v;
  }
  for (int off = 16; off > 0; off >>= 1) {
    s  += __shfl_down(s, off, 32);
    s2 += __shfl_down(s2, off, 32);
  }
  int lane = threadIdx.x & 31, wid = threadIdx.x >> 5;
  if (lane == 0) { rs[wid] = s; rs2[wid] = s2; }
  __syncthreads();
  if (threadIdx.x == 0) {
    float ts = 0.f, ts2 = 0.f;
    int nw = blockDim.x >> 5;
    for (int i = 0; i < nw; ++i) { ts += rs[i]; ts2 += rs2[i]; }
    rs[0] = ts; rs2[0] = ts2;
  }
  __syncthreads();
  float mean = rs[0] * (1.0f / HID);
  float var  = rs2[0] * (1.0f / HID) - mean * mean;
  float inv  = rsqrtf(var + 1e-7f);
  float* o = out + (size_t)row * HID;
  for (int c = threadIdx.x; c < HID; c += blockDim.x)
    o[c] = (x[c] - mean) * inv * gamma[c] + beta[c];
}

// ---------------------------------------------------------------------------
// Host launch
// ---------------------------------------------------------------------------
extern "C" void kernel_launch(void* const* d_in, const int* in_sizes, int n_in,
                              void* d_out, int out_size, void* d_ws,
                              size_t ws_size, hipStream_t stream) {
  (void)in_sizes; (void)n_in; (void)out_size; (void)ws_size;
  const float* hidden = (const float*)d_in[0];
  /* d_in[1] attention_mask: unused (faithful to reference) */
  const float* rel   = (const float*)d_in[2];
  const float* Wq    = (const float*)d_in[3];
  const float* bq    = (const float*)d_in[4];
  const float* Wk    = (const float*)d_in[5];
  const float* bk    = (const float*)d_in[6];
  const float* Wv    = (const float*)d_in[7];
  const float* bv    = (const float*)d_in[8];
  const float* Wo    = (const float*)d_in[9];
  const float* bo    = (const float*)d_in[10];
  const float* gamma = (const float*)d_in[11];
  const float* beta  = (const float*)d_in[12];
  float* out = (float*)d_out;

  char* ws = (char*)d_ws;
  size_t off = 0;
  auto take = [&](size_t bytes) -> char* {
    char* p = ws + off;
    off = (off + bytes + 255) & ~(size_t)255;
    return p;
  };
  const size_t BSH = (size_t)NB * SEQ * HID;
  int* tbl_c2p     = (int*)take(TBL * 4);
  int* tbl_p2c     = (int*)take(TBL * 4);
  _Float16* h16    = (_Float16*)take(BSH * 2);
  _Float16* wq16   = (_Float16*)take((size_t)HID * HID * 2);
  _Float16* wk16   = (_Float16*)take((size_t)HID * HID * 2);
  _Float16* wv16   = (_Float16*)take((size_t)HID * HID * 2);
  _Float16* wo16   = (_Float16*)take((size_t)HID * HID * 2);
  _Float16* re16   = (_Float16*)take((size_t)P2 * HID * 2);
  _Float16* q16    = (_Float16*)take(BSH * 2);
  _Float16* k16    = (_Float16*)take(BSH * 2);
  _Float16* v16    = (_Float16*)take(BSH * 2);
  _Float16* posk16 = (_Float16*)take((size_t)P2 * HID * 2);
  _Float16* p2c16  = (_Float16*)take((size_t)NB * NH * SEQ * P2 * 2);
  _Float16* ctx16  = (_Float16*)take(BSH * 2);
  float* preln     = (float*)take(BSH * 4);

  build_tables_kernel<<<dim3((TBL + 255) / 256), dim3(256), 0, stream>>>(
      tbl_c2p, tbl_p2c);

  cvt_f32_f16_kernel<<<dim3(1024), dim3(256), 0, stream>>>(hidden, h16, (int)BSH);
  cvt_f32_f16_kernel<<<dim3(512), dim3(256), 0, stream>>>(Wq, wq16, HID * HID);
  cvt_f32_f16_kernel<<<dim3(512), dim3(256), 0, stream>>>(Wk, wk16, HID * HID);
  cvt_f32_f16_kernel<<<dim3(512), dim3(256), 0, stream>>>(Wv, wv16, HID * HID);
  cvt_f32_f16_kernel<<<dim3(512), dim3(256), 0, stream>>>(Wo, wo16, HID * HID);
  cvt_f32_f16_kernel<<<dim3(256), dim3(256), 0, stream>>>(rel, re16, P2 * HID);

  dim3 blk(128);
  gemm_bias_f16_kernel<<<dim3(NB * SEQ / 64, HID / 64), blk, 0, stream>>>(
      h16, wq16, bq, q16, NB * SEQ, HID, HID);
  gemm_bias_f16_kernel<<<dim3(NB * SEQ / 64, HID / 64), blk, 0, stream>>>(
      h16, wk16, bk, k16, NB * SEQ, HID, HID);
  gemm_bias_f16_kernel<<<dim3(NB * SEQ / 64, HID / 64), blk, 0, stream>>>(
      h16, wv16, bv, v16, NB * SEQ, HID, HID);
  gemm_bias_f16_kernel<<<dim3(P2 / 64, HID / 64), blk, 0, stream>>>(
      re16, wk16, bk, posk16, P2, HID, HID);

  p2c_att_kernel<<<dim3(SEQ / 32, P2 / 64, NB * NH), blk, 0, stream>>>(
      k16, posk16, p2c16);

  const size_t smem = 222336;  // see layout above
  (void)hipFuncSetAttribute((const void*)attn_kernel,
                            hipFuncAttributeMaxDynamicSharedMemorySize,
                            (int)smem);
  attn_kernel<<<dim3(SEQ / 32, NH, NB), blk, smem, stream>>>(
      q16, k16, v16, posk16, p2c16, tbl_c2p, tbl_p2c, ctx16);

  gemm_out_res_kernel<<<dim3(NB * SEQ / 64, HID / 64), blk, 0, stream>>>(
      ctx16, wo16, bo, hidden, preln, NB * SEQ, HID, HID);
  ln_kernel<<<dim3(NB * SEQ), dim3(256), 0, stream>>>(preln, gamma, beta, out);
}
